// NeuralHMM_77197742178377
// MI455X (gfx1250) — compile-verified, matrix operand order checked by reading, not run on hardware
//
#include <hip/hip_runtime.h>
#include <cstddef>
#include <cstdint>

// ---------------- types ----------------
typedef unsigned short u16;
typedef unsigned int   u32;
typedef __attribute__((ext_vector_type(16))) __bf16 v16bf;
typedef __attribute__((ext_vector_type(8)))  float  v8f;
typedef __attribute__((ext_vector_type(4)))  unsigned int u32x4;
typedef __attribute__((ext_vector_type(8)))  int  i32x8;
typedef __attribute__((ext_vector_type(4)))  int  i32x4;

union FragU { v16bf v; uint4 u[2]; };

// problem sizes
#define BN  4
#define TT  2048
#define BT  8192      // B*T
#define DD  80
#define SS  64
#define HH  256
#define CC  128

// Tensor Data Mover availability (ROCm7.2 clang-22 = 5-arg; clang-23 = 6-arg)
#if defined(__has_builtin)
#if __has_builtin(__builtin_amdgcn_tensor_load_to_lds) && \
    __has_builtin(__builtin_amdgcn_s_wait_tensorcnt)
#define USE_TDM 1
#endif
#endif
#ifndef USE_TDM
#define USE_TDM 0
#endif

__device__ __forceinline__ u16 f2bf(float f) {
    u32 u = __float_as_uint(f);
    u32 r = (u + 0x7FFFu + ((u >> 16) & 1u)) >> 16;
    return (u16)r;
}

// ---------------- repack fp32 -> bf16 with K padding ----------------
__global__ void convert_pad(const float* __restrict__ src, int srcLd,
                            u16* __restrict__ dst, int rows, int kcopy, int kdst) {
    int idx = blockIdx.x * blockDim.x + threadIdx.x;
    int total = rows * kdst;
    if (idx >= total) return;
    int r = idx / kdst;
    int k = idx - r * kdst;
    dst[idx] = (k < kcopy) ? f2bf(src[(size_t)r * srcLd + k]) : (u16)0;
}

__global__ void kzero16(u16* __restrict__ p, int n) {
    int i = blockIdx.x * blockDim.x + threadIdx.x;
    if (i < n) p[i] = 0;
}

// tb1' = tb1 + (1/64) * sum_{c=128..191} tw1[h, c]   (uniform concat folded into bias)
__global__ void ktb1(const float* __restrict__ tw1, const float* __restrict__ tb1,
                     float* __restrict__ tb1p) {
    int h = threadIdx.x;
    if (h < HH) {
        float acc = 0.f;
        for (int c = CC; c < CC + SS; ++c) acc += tw1[h * (CC + SS) + c];
        tb1p[h] = tb1[h] + acc * (1.0f / 64.0f);
    }
}

// concat bias for the fused mean/logvar GEMM: [mb | lb | 0]
__global__ void kbias192(const float* __restrict__ mb, const float* __restrict__ lb,
                         float* __restrict__ mbc) {
    int i = threadIdx.x;  // 192 threads
    if (i < DD)            mbc[i] = mb[i];
    else if (i < 2 * DD)   mbc[i] = lb[i - DD];
    else                   mbc[i] = 0.f;
}

// Per-state precompute: off_mean/off_lv via small dot products, then pack the
// B-matrix of the mahalanobis GEMM: [E2 | -2*m*E2 | m^2*E2 | 0pad], + sumOL.
__global__ void koff(const float* __restrict__ emb, const float* __restrict__ mw,
                     const float* __restrict__ lw, u16* __restrict__ Bp,
                     float* __restrict__ sumOL) {
    int s = blockIdx.x, tid = threadIdx.x;   // block 128
    __shared__ float red[128];
    float olv = 0.f;
    if (tid < DD) {
        float om = 0.f, ol = 0.f;
        for (int h = 0; h < HH; ++h) {
            float e = emb[s * HH + h];
            om += e * mw[tid * HH + h];
            ol += e * lw[tid * HH + h];
        }
        float E2 = __expf(-ol);
        size_t o = (size_t)s * 256;
        Bp[o + tid]            = f2bf(E2);
        Bp[o + DD + tid]       = f2bf(-2.0f * om * E2);
        Bp[o + 2 * DD + tid]   = f2bf(om * om * E2);
        olv = ol;
    } else if (tid < DD + 16) {
        Bp[(size_t)s * 256 + 3 * DD + (tid - DD)] = 0;  // pad cols 240..255
    }
    red[tid] = olv;
    __syncthreads();
    for (int off = 64; off > 0; off >>= 1) {
        if (tid < off) red[tid] += red[tid + off];
        __syncthreads();
    }
    if (tid == 0) sumOL[s] = red[0];
}

// Pack A-matrix of the mahalanobis GEMM: [r^2*E1 | r*E1 | E1 | 0pad], + sumBL.
__global__ void kpackA(const float* __restrict__ obs, const float* __restrict__ baseo,
                       u16* __restrict__ Ap, float* __restrict__ sumBL) {
    int bt = blockIdx.x, tid = threadIdx.x;  // block 128
    __shared__ float red[128];
    float blv = 0.f;
    if (tid < DD) {
        float r = obs[(size_t)bt * DD + tid] - baseo[(size_t)bt * 192 + tid];
        blv = baseo[(size_t)bt * 192 + DD + tid];
        float E1 = __expf(-blv);
        size_t o = (size_t)bt * 256;
        Ap[o + tid]          = f2bf(r * r * E1);
        Ap[o + DD + tid]     = f2bf(r * E1);
        Ap[o + 2 * DD + tid] = f2bf(E1);
    } else if (tid < DD + 16) {
        Ap[(size_t)bt * 256 + 3 * DD + (tid - DD)] = 0;
    }
    red[tid] = blv;
    __syncthreads();
    for (int off = 64; off > 0; off >>= 1) {
        if (tid < off) red[tid] += red[tid + off];
        __syncthreads();
    }
    if (tid == 0) sumBL[bt] = red[0];
}

// ---------------- bf16 WMMA GEMM: C[M,N] = epilogue(A[M,K] @ B[N,K]^T) ----------------
// MODE 0: outF = acc + bias[col]                     (fp32 out)
// MODE 1: outB = bf16(relu(acc + bias[col]))         (bf16 out, feeds next GEMM)
// MODE 2: outF = -0.5*(cadd + rowAdd[row] + colAdd[col] + acc)   (obs_lp epilogue)
// Block: 128 threads = 4 waves; wave -> 32x32 C tile (2x2 WMMA); block -> 64x64.
template <int MODE>
__global__ void __launch_bounds__(128) gemm_bf16(
    const u16* __restrict__ A, const u16* __restrict__ Bw,
    const float* __restrict__ bias, const float* __restrict__ rowAdd,
    const float* __restrict__ colAdd, float cadd,
    float* __restrict__ outF, u16* __restrict__ outB, int K, int ldc) {

    const int lane = threadIdx.x & 31;
    const int wave = threadIdx.x >> 5;
    const int waveRow = wave >> 1, waveCol = wave & 1;
    const int tileM = blockIdx.x * 64, tileN = blockIdx.y * 64;
    const int l15 = lane & 15, half = lane >> 4;

    const v8f vzero = {0.f, 0.f, 0.f, 0.f, 0.f, 0.f, 0.f, 0.f};
    v8f acc[2][2];
    acc[0][0] = vzero; acc[0][1] = vzero; acc[1][0] = vzero; acc[1][1] = vzero;

    const u16* aBase = A  + (size_t)(tileM + waveRow * 32 + l15) * K;
    const u16* bBase = Bw + (size_t)(tileN + waveCol * 32 + l15) * K;

    for (int kk = 0; kk < K; kk += 32) {
        FragU a0, a1, b0, b1;
        const u16* pa0 = aBase + kk + half * 8;
        const u16* pa1 = pa0 + (size_t)16 * K;
        const u16* pb0 = bBase + kk + half * 8;
        const u16* pb1 = pb0 + (size_t)16 * K;
        a0.u[0] = *reinterpret_cast<const uint4*>(pa0);
        a0.u[1] = *reinterpret_cast<const uint4*>(pa0 + 16);
        a1.u[0] = *reinterpret_cast<const uint4*>(pa1);
        a1.u[1] = *reinterpret_cast<const uint4*>(pa1 + 16);
        b0.u[0] = *reinterpret_cast<const uint4*>(pb0);
        b0.u[1] = *reinterpret_cast<const uint4*>(pb0 + 16);
        b1.u[0] = *reinterpret_cast<const uint4*>(pb1);
        b1.u[1] = *reinterpret_cast<const uint4*>(pb1 + 16);
        acc[0][0] = __builtin_amdgcn_wmma_f32_16x16x32_bf16(false, a0.v, false, b0.v,
                                                            (short)0, acc[0][0], false, false);
        acc[0][1] = __builtin_amdgcn_wmma_f32_16x16x32_bf16(false, a0.v, false, b1.v,
                                                            (short)0, acc[0][1], false, false);
        acc[1][0] = __builtin_amdgcn_wmma_f32_16x16x32_bf16(false, a1.v, false, b0.v,
                                                            (short)0, acc[1][0], false, false);
        acc[1][1] = __builtin_amdgcn_wmma_f32_16x16x32_bf16(false, a1.v, false, b1.v,
                                                            (short)0, acc[1][1], false, false);
    }

#pragma unroll
    for (int i = 0; i < 2; ++i) {
        int rbase = tileM + waveRow * 32 + i * 16 + half * 8;
#pragma unroll
        for (int j = 0; j < 2; ++j) {
            int col = tileN + waveCol * 32 + j * 16 + l15;
#pragma unroll
            for (int v = 0; v < 8; ++v) {
                int row = rbase + v;
                float x = acc[i][j][v];
                size_t o = (size_t)row * ldc + col;
                if constexpr (MODE == 0) {
                    outF[o] = x + bias[col];
                } else if constexpr (MODE == 1) {
                    outB[o] = f2bf(fmaxf(x + bias[col], 0.0f));
                } else {
                    outF[o] = -0.5f * (cadd + rowAdd[row] + colAdd[col] + x);
                }
            }
        }
    }
}

// ---------------- TDM: async 64x64 fp32 tile -> LDS, with HW row padding ----------------
#if USE_TDM
__device__ __forceinline__ void tdm_load_tile(const float* gsrc, unsigned int ldsAddr) {
    // D# group 0: count=1 | lds_addr | global_addr[56:0] | type=2
    union { u32x4 v; unsigned int a[4]; } g0;
    unsigned long long ga = (unsigned long long)(uintptr_t)gsrc;
    g0.a[0] = 1u;                                  // count=1 (valid user descriptor)
    g0.a[1] = ldsAddr;                             // lds_addr (bytes)
    g0.a[2] = (unsigned int)ga;                    // global_addr[31:0]
    g0.a[3] = (unsigned int)((ga >> 32) & 0x01FFFFFFull) | (2u << 30);  // addr[56:32]|type=2
    // D# group 1: data_size=4B, pad every 64 DWORDs by 1 DWORD (-> LDS stride 65),
    // 1-D tile: tile_dim0 = tensor_dim0 = 4096 elements, stride 4096.
    union { i32x8 v; int a[8]; } g1;
    g1.a[0] = (int)((2u << 16) | (1u << 20) | (5u << 22)); // data_size|pad_en|interval=64DW
    g1.a[1] = (int)(4096u << 16);                  // tensor_dim0[15:0] -> bits[31:16]
    g1.a[2] = (int)(1u << 16);                     // tensor_dim0 hi=0; tensor_dim1=1
    g1.a[3] = (int)(4096u << 16);                  // tensor_dim1 hi=0; tile_dim0=4096
    g1.a[4] = 0;                                   // tile_dim1=0 (unused), tile_dim2=0
    g1.a[5] = 4096;                                // tensor_dim0_stride lo
    g1.a[6] = 0;                                   // stride hi / tensor_dim1_stride lo
    g1.a[7] = 0;
    i32x4 zz = {0, 0, 0, 0};
#if defined(__clang_major__) && (__clang_major__ >= 23)
    union { i32x8 v; int a[8]; } z8;
    for (int i = 0; i < 8; ++i) z8.a[i] = 0;
    __builtin_amdgcn_tensor_load_to_lds(g0.v, g1.v, zz, zz, z8.v, 0);
#else
    __builtin_amdgcn_tensor_load_to_lds(g0.v, g1.v, zz, zz, 0);
#endif
}
#endif

// ---------------- HMM forward scan (log domain), log-softmax fused ----------------
// One block per batch, 64 threads (one per state). Per step: 64x64 logits tile
// staged in LDS (TDM double-buffered when available), per-row LSE in LDS, then
// logsumexp mat-vec for the alpha recurrence.
__global__ void __launch_bounds__(64) kscan(const float* __restrict__ logits,
                                            const float* __restrict__ obslp,
                                            float* __restrict__ out) {
    int b = blockIdx.x, j = threadIdx.x;
    __shared__ float alpha[64];
    __shared__ float tileS[2][64 * 65];   // 65-float row stride (bank-conflict free)
    __shared__ float rowLse[64];
    __shared__ float red[64];

    alpha[j] = -4.1588830833596715f /* -log(64) */ + obslp[(size_t)(b * TT) * SS + j];
    __syncthreads();

#if USE_TDM
    if (threadIdx.x < 32) {   // wave 0 owns the DMA pipeline (TENSORcnt is per-wave)
        tdm_load_tile(logits + (size_t)(b * TT + 1) * (SS * SS),
                      (unsigned int)(uintptr_t)&tileS[0][0]);
    }
#endif

    for (int t = 1; t < TT; ++t) {
        const int p = (t - 1) & 1;
        const float* lt = logits + (size_t)(b * TT + t) * (SS * SS);

#if USE_TDM
        if (threadIdx.x < 32) __builtin_amdgcn_s_wait_tensorcnt(0);
        __syncthreads();      // tile[p] ready for everyone; prior reads of tile[t&1] done
        if (t + 1 < TT && threadIdx.x < 32) {
            tdm_load_tile(logits + (size_t)(b * TT + t + 1) * (SS * SS),
                          (unsigned int)(uintptr_t)&tileS[t & 1][0]);
        }
#else
        if (t + 1 < TT) {     // pull next step's tile toward the caches
            const float* nxt = lt + SS * SS;
            __builtin_prefetch(nxt + j * 64, 0, 0);
            __builtin_prefetch(nxt + j * 64 + 32, 0, 0);
        }
        for (int idx = j; idx < SS * SS; idx += 64)
            tileS[p][(idx >> 6) * 65 + (idx & 63)] = lt[idx];
        __syncthreads();
#endif
        const float* tile = &tileS[p][0];

        {   // per-previous-state row LSE (log-softmax denominator), thread j = row j
            float m = -3.0e38f, s = 0.f;
            for (int k = 0; k < SS; ++k) {
                float v = tile[j * 65 + k];
                if (v > m) { s = s * __expf(m - v) + 1.f; m = v; }
                else       { s += __expf(v - m); }
            }
            rowLse[j] = m + __logf(s);
        }
        __syncthreads();

        float m = -3.0e38f, s = 0.f;
        for (int i = 0; i < SS; ++i) {
            float v = alpha[i] + tile[i * 65 + j] - rowLse[i];
            if (v > m) { s = s * __expf(m - v) + 1.f; m = v; }
            else       { s += __expf(v - m); }
        }
        float na = m + __logf(s) + obslp[(size_t)(b * TT + t) * SS + j];
        __syncthreads();
        alpha[j] = na;
        __syncthreads();
    }

    float a = alpha[j];
    red[j] = a; __syncthreads();
    for (int off = 32; off > 0; off >>= 1) {
        if (j < off) red[j] = fmaxf(red[j], red[j + off]);
        __syncthreads();
    }
    float mx = red[0]; __syncthreads();
    red[j] = __expf(a - mx); __syncthreads();
    for (int off = 32; off > 0; off >>= 1) {
        if (j < off) red[j] += red[j + off];
        __syncthreads();
    }
    if (j == 0) out[b] = mx + __logf(red[0]);
}

// ---------------- host-side orchestration ----------------
static inline int cdiv(int a, int b) { return (a + b - 1) / b; }

extern "C" void kernel_launch(void* const* d_in, const int* in_sizes, int n_in,
                              void* d_out, int out_size, void* d_ws, size_t ws_size,
                              hipStream_t stream) {
    (void)in_sizes; (void)n_in; (void)out_size; (void)ws_size;
    const float* obs = (const float*)d_in[0];
    const float* ctx = (const float*)d_in[1];
    const float* emb = (const float*)d_in[2];
    const float* fw1 = (const float*)d_in[3];
    const float* fb1 = (const float*)d_in[4];
    const float* fw2 = (const float*)d_in[5];
    const float* fb2 = (const float*)d_in[6];
    const float* mw  = (const float*)d_in[7];
    const float* mb  = (const float*)d_in[8];
    const float* lw  = (const float*)d_in[9];
    const float* lb  = (const float*)d_in[10];
    const float* tw1 = (const float*)d_in[11];
    const float* tb1 = (const float*)d_in[12];
    const float* tw2 = (const float*)d_in[13];
    const float* tb2 = (const float*)d_in[14];
    const float* tw3 = (const float*)d_in[15];
    const float* tb3 = (const float*)d_in[16];
    float* out = (float*)d_out;

    // workspace carve-up (~170 MB; logits tensor alone is 134 MB and fits in L2)
    char* ws = (char*)d_ws;
    size_t off = 0;
    auto alloc = [&](size_t bytes) -> void* {
        void* p = ws + off;
        off = (off + bytes + 255) & ~(size_t)255;
        return p;
    };
    u16*   obsb   = (u16*)  alloc((size_t)BT * 96 * 2);
    u16*   ctxb   = (u16*)  alloc((size_t)BT * 128 * 2);
    u16*   fw1b   = (u16*)  alloc((size_t)HH * 96 * 2);
    u16*   fw2b   = (u16*)  alloc((size_t)HH * 256 * 2);
    u16*   mwlwb  = (u16*)  alloc((size_t)192 * 256 * 2);
    u16*   tw1b   = (u16*)  alloc((size_t)HH * 128 * 2);
    u16*   tw2b   = (u16*)  alloc((size_t)HH * 256 * 2);
    u16*   tw3b   = (u16*)  alloc((size_t)4096 * 256 * 2);
    float* tb1p   = (float*)alloc((size_t)HH * 4);
    float* mbc    = (float*)alloc((size_t)192 * 4);
    u16*   f1b    = (u16*)  alloc((size_t)BT * 256 * 2);
    u16*   f2b    = (u16*)  alloc((size_t)BT * 256 * 2);
    float* baseo  = (float*)alloc((size_t)BT * 192 * 4);
    u16*   Ap     = (u16*)  alloc((size_t)BT * 256 * 2);
    u16*   Bp     = (u16*)  alloc((size_t)SS * 256 * 2);
    float* sumBL  = (float*)alloc((size_t)BT * 4);
    float* sumOL  = (float*)alloc((size_t)SS * 4);
    float* obslp  = (float*)alloc((size_t)BT * SS * 4);
    u16*   h1b    = (u16*)  alloc((size_t)BT * 256 * 2);
    u16*   h2b    = (u16*)  alloc((size_t)BT * 256 * 2);
    float* logits = (float*)alloc((size_t)BT * 4096 * 4);

    // 1) repack fp32 -> bf16 (K padded to multiples of 32)
    auto conv = [&](const float* src, int srcLd, u16* dst, int rows, int kcopy, int kdst) {
        int total = rows * kdst;
        convert_pad<<<cdiv(total, 256), 256, 0, stream>>>(src, srcLd, dst, rows, kcopy, kdst);
    };
    conv(obs, DD, obsb, BT, DD, 96);
    conv(ctx, CC, ctxb, BT, CC, 128);
    conv(fw1, DD, fw1b, HH, DD, 96);
    conv(fw2, HH, fw2b, HH, HH, 256);
    conv(mw,  HH, mwlwb,            DD, HH, 256);   // rows 0..79
    conv(lw,  HH, mwlwb + 80 * 256, DD, HH, 256);   // rows 80..159
    kzero16<<<cdiv(32 * 256, 256), 256, 0, stream>>>(mwlwb + 160 * 256, 32 * 256);
    conv(tw1, CC + SS, tw1b, HH, CC, 128);          // uniform part folded into bias
    conv(tw2, HH, tw2b, HH, HH, 256);
    conv(tw3, HH, tw3b, 4096, HH, 256);

    // 2) small precomputes
    koff<<<SS, 128, 0, stream>>>(emb, mw, lw, Bp, sumOL);
    ktb1<<<1, 256, 0, stream>>>(tw1, tb1, tb1p);
    kbias192<<<1, 192, 0, stream>>>(mb, lb, mbc);

    dim3 blk(128);
    // 3) observation MLP: f1 = relu(obs @ fw1^T + fb1); f2 = relu(f1 @ fw2^T + fb2)
    gemm_bf16<1><<<dim3(BT / 64, 256 / 64), blk, 0, stream>>>(
        obsb, fw1b, fb1, nullptr, nullptr, 0.f, nullptr, f1b, 96, 256);
    gemm_bf16<1><<<dim3(BT / 64, 256 / 64), blk, 0, stream>>>(
        f1b, fw2b, fb2, nullptr, nullptr, 0.f, nullptr, f2b, 256, 256);
    // 4) fused base_mean|base_lv GEMM: (8192 x 256) @ (192 x 256)^T
    gemm_bf16<0><<<dim3(BT / 64, 192 / 64), blk, 0, stream>>>(
        f2b, mwlwb, mbc, nullptr, nullptr, 0.f, baseo, nullptr, 256, 192);
    // 5) pack mahalanobis operands, then obs_lp GEMM with fused gaussian epilogue
    kpackA<<<BT, 128, 0, stream>>>(obs, baseo, Ap, sumBL);
    const float cadd = 80.0f * 1.8378770664093453f;  // D * log(2*pi)
    gemm_bf16<2><<<dim3(BT / 64, SS / 64), blk, 0, stream>>>(
        Ap, Bp, nullptr, sumBL, sumOL, cadd, obslp, nullptr, 256, SS);
    // 6) transition MLP (uniform concat folded into tb1p)
    gemm_bf16<1><<<dim3(BT / 64, 256 / 64), blk, 0, stream>>>(
        ctxb, tw1b, tb1p, nullptr, nullptr, 0.f, nullptr, h1b, 128, 256);
    gemm_bf16<1><<<dim3(BT / 64, 256 / 64), blk, 0, stream>>>(
        h1b, tw2b, tb2, nullptr, nullptr, 0.f, nullptr, h2b, 256, 256);
    // 7) logits GEMM (the big one: 8192 x 4096 x 256)
    gemm_bf16<0><<<dim3(BT / 64, 4096 / 64), blk, 0, stream>>>(
        h2b, tw3b, tb3, nullptr, nullptr, 0.f, logits, nullptr, 256, 4096);
    // 8) HMM forward scan with fused log-softmax; writes log-likelihood (B,)
    kscan<<<BN, 64, 0, stream>>>(logits, obslp, out);
}
